// TRGNet_49452253446609
// MI455X (gfx1250) — matrix operand order repaired and checked by reading, not compiled, not run on hardware
//
#include <hip/hip_runtime.h>
#include <hip/hip_bf16.h>
#include <math.h>

typedef unsigned short ushort_t;
typedef unsigned int   uint_t;
typedef __attribute__((ext_vector_type(16))) __bf16 v16bf;
typedef __attribute__((ext_vector_type(8)))  float  v8f;

#define B_N   16
#define C_IN  48
#define C_OUT 48
#define CPAD  64                      // channels padded to 64 (16 zero lanes)
#define PSTRIDE 72                    // LDS pixel stride (ushorts): 144B = 36 banks
                                      // -> conflict-free ds_load_b128 across lanes
#define HH    128
#define WW    128
#define SP    7
#define KB_N  98                      // 64*49 / 32  -> exact, no tail
#define TCOLS (WW + 6)                // 134 halo cols
#define ROWS_PB 2                     // output rows per block
#define LROWS (SP + ROWS_PB - 1)      // 8 halo rows in LDS
#define LDSX_ELTS (LROWS * TCOLS * PSTRIDE)  // 8*134*72 = 77184 bf16 = 154368 B

__device__ __forceinline__ ushort_t f2bf(float f) {
  union { float f; uint_t u; } x; x.f = f;
  uint_t r = x.u + 0x7FFFu + ((x.u >> 16) & 1u);  // round-to-nearest-even
  return (ushort_t)(r >> 16);
}

// ---------------------------------------------------------------------------
// Kernel 0: NCHW f32 -> NHWC(64) bf16, channels 48..63 zero.
// ---------------------------------------------------------------------------
__global__ void k_convert(const float* __restrict__ x, ushort_t* __restrict__ xbf) {
  int idx = blockIdx.x * blockDim.x + threadIdx.x;   // 0 .. 16,777,215
  int m = idx & 63;  int t = idx >> 6;
  int w = t & 127;   t >>= 7;
  int h = t & 127;   int b = t >> 7;
  float v = 0.0f;
  if (m < C_IN) v = x[(((size_t)b * C_IN + m) * HH + h) * WW + w];
  xbf[idx] = f2bf(v);
}

// ---------------------------------------------------------------------------
// Kernel 1: basis (49x98) per batch -> contract with weights -> pack filters
// into the WMMA 16x32 bf16 A-fragment layout with K-order k = m + 64*r.
// ckA layout: [b][cot][kb 0..97][lane 0..31][slot 0..15] bf16.
//   khalf = (kk>>3)&1; slot = (kk&7) + 8*(kk>=16); lane = 16*khalf + (n%16)
// Entire slice zeroed first so padded m in [48,64) stays 0.
// ---------------------------------------------------------------------------
__global__ void k_filters(const float* __restrict__ theta,
                          const float* __restrict__ weights,
                          ushort_t* __restrict__ ckA) {
  __shared__ float basisS[49 * 98];
  __shared__ float wn[98 * C_IN];
  const int tid = threadIdx.x;
  const int b = blockIdx.x, cot = blockIdx.y;
  const float th = theta[b];
  const float ct = cosf(th), st = sinf(th);
  const float vfac = (float)(M_PI / SP * (SP - 1));

  ushort_t* slice = ckA + (size_t)(b * 3 + cot) * KB_N * 512;

  // zero the whole slice (98*512 ushorts = 6272 uint4)
  for (int idx = tid; idx < KB_N * 512 / 8; idx += blockDim.x)
    ((uint4*)slice)[idx] = make_uint4(0u, 0u, 0u, 0u);

  // basis
  for (int idx = tid; idx < 49 * 98; idx += blockDim.x) {
    int r = idx / 98, k2 = idx % 98;
    int i = r / SP, j = r % SP;
    float X = (float)(j - 3) * (1.0f / 3.0f);
    float Y = (float)(i - 3) * (1.0f / 3.0f);
    float Xr = ct * X - st * Y;
    float Yr = st * X + ct * Y;
    float mask = __expf(-fmaxf(Xr * Xr + Yr * Yr - 1.0f, 0.0f) * 5.0f);
    int kq = k2 % 49;
    int kr = kq / SP, lr = kq % SP;
    float kk = (float)(kr >= 4 ? kr - SP : kr);
    float ll = (float)(lr >= 4 ? lr - SP : lr);
    float ang = kk * vfac * Xr + ll * vfac * Yr;
    basisS[idx] = (k2 < 49 ? cosf(ang) : sinf(ang)) * mask;
  }
  __syncthreads();

  for (int nl = 0; nl < 16; ++nl) {
    int n = cot * 16 + nl;
    for (int idx = tid; idx < 98 * C_IN; idx += blockDim.x)
      wn[idx] = weights[(size_t)n * 98 * C_IN + idx];
    __syncthreads();
    for (int idx = tid; idx < C_IN * 49; idx += blockDim.x) {
      int m = idx % C_IN, r = idx / C_IN;
      float acc = 0.0f;
      #pragma unroll 7
      for (int k2 = 0; k2 < 98; ++k2)
        acc += basisS[r * 98 + k2] * wn[k2 * C_IN + m];
      int k    = m + CPAD * r;
      int kb   = k >> 5, kk = k & 31;
      int half = (kk >> 3) & 1;
      int slot = (kk & 7) + ((kk >> 4) << 3);
      int lane = half * 16 + nl;
      slice[((size_t)kb * 32 + lane) * 16 + slot] = f2bf(acc);
    }
    __syncthreads();
  }
}

// ---------------------------------------------------------------------------
// Kernel 2: implicit-GEMM grouped conv, v_wmma_f32_16x16x32_bf16.
// Block = (2 output rows, cot, b); 8 waves, each wave owns one 16-pixel
// column tile on BOTH rows (two accumulators) -> A fetch amortized 2x.
// Halo tile [8 rows][134 cols][72-stride] bf16 staged once in LDS (pixel
// stride 144B = 36 banks -> bank-conflict-free ds_load_b128).
// Steady state per kb: 2 global_load_b128 (A) + 4 ds_load_b128 (B) + 2 wmma.
// ---------------------------------------------------------------------------
union AFrag { v16bf v; uint4 q[2]; };
union BFrag { v16bf v; uint4 q[2]; };

__global__ void k_conv(const ushort_t* __restrict__ xbf,
                       const ushort_t* __restrict__ ckA,
                       const float* __restrict__ cbias,
                       float* __restrict__ out) {
  extern __shared__ ushort_t ldsx[];   // [8 rows][134 cols][72] bf16
  const int tid = threadIdx.x;
  const int h0  = blockIdx.x * ROWS_PB;
  const int cot = blockIdx.y;
  const int b   = blockIdx.z;

  // stage halo tile: rows h0-3 .. h0+4, cols -3..130, 64 ch (128B/pixel)
  for (int idx = tid; idx < LROWS * TCOLS * 8; idx += blockDim.x) {
    int pix = idx >> 3, ch8 = idx & 7;          // 8 x (8 channels) per pixel
    int row = pix / TCOLS, col = pix % TCOLS;
    int h = h0 + row - 3, w = col - 3;
    uint4 val = make_uint4(0u, 0u, 0u, 0u);
    if (h >= 0 && h < HH && w >= 0 && w < WW)
      val = *(const uint4*)(xbf + (((size_t)b * HH + h) * WW + w) * CPAD + ch8 * 8);
    *(uint4*)(ldsx + (size_t)pix * PSTRIDE + ch8 * 8) = val;
  }
  __syncthreads();

  const int lane  = tid & 31;
  const int wv    = tid >> 5;
  const int npix  = wv * 16 + (lane & 15);        // output column (N index)
  const int lhalf = lane >> 4;
  const int laneoff = npix * PSTRIDE + lhalf * 8;  // per-lane, kb-invariant part

  const uint4* aP = (const uint4*)ckA + (size_t)(b * 3 + cot) * KB_N * 64;
  v8f acc0 = {};
  v8f acc1 = {};

  #pragma unroll 2
  for (int kb = 0; kb < KB_N; ++kb) {
    int r    = kb >> 1;                           // uniform (SALU)
    int rr   = r / SP;
    int rc   = r - rr * SP;
    int base = (rr * TCOLS + rc) * PSTRIDE + ((kb & 1) << 5);  // uniform
    int off  = base + laneoff;                    // ushort index, 16B aligned

    AFrag a;
    a.q[0] = aP[kb * 64 + lane * 2];
    a.q[1] = aP[kb * 64 + lane * 2 + 1];
    __builtin_prefetch((const void*)(aP + ((kb + 8 < KB_N ? kb + 8 : KB_N - 1) * 64 + lane * 2)), 0, 1);

    BFrag bf0, bf1;                               // row h0 and row h0+1 tiles
    bf0.q[0] = *(const uint4*)(ldsx + off);
    bf0.q[1] = *(const uint4*)(ldsx + off + 16);
    bf1.q[0] = *(const uint4*)(ldsx + off + TCOLS * PSTRIDE);   // +1 halo row
    bf1.q[1] = *(const uint4*)(ldsx + off + TCOLS * PSTRIDE + 16);

    acc0 = __builtin_amdgcn_wmma_f32_16x16x32_bf16(
        false, a.v, false, bf0.v, (short)0, acc0, false, false);
    acc1 = __builtin_amdgcn_wmma_f32_16x16x32_bf16(
        false, a.v, false, bf1.v, (short)0, acc1, false, false);
  }

  const int w = wv * 16 + (lane & 15);
  #pragma unroll
  for (int p = 0; p < 8; ++p) {
    int n = cot * 16 + p + 8 * lhalf;
    float cb = cbias[n];
    size_t o = (((size_t)b * C_OUT + n) * HH + h0) * WW + w;
    out[o]      = acc0[p] + cb;
    out[o + WW] = acc1[p] + cb;
  }
}

// ---------------------------------------------------------------------------
extern "C" void kernel_launch(void* const* d_in, const int* in_sizes, int n_in,
                              void* d_out, int out_size, void* d_ws, size_t ws_size,
                              hipStream_t stream) {
  const float* x       = (const float*)d_in[0];
  const float* theta   = (const float*)d_in[1];
  const float* weights = (const float*)d_in[2];
  const float* cbias   = (const float*)d_in[3];
  float* out = (float*)d_out;

  ushort_t* xbf = (ushort_t*)d_ws;                                 // 33.5 MB
  ushort_t* ckA = xbf + (size_t)B_N * HH * WW * CPAD;              // 4.8 MB

  int totalx = B_N * HH * WW * CPAD;                               // 16,777,216
  k_convert<<<totalx / 256, 256, 0, stream>>>(x, xbf);
  k_filters<<<dim3(B_N, 3), 256, 0, stream>>>(theta, weights, ckA);
  k_conv<<<dim3(HH / ROWS_PB, 3, B_N), 256, LDSX_ELTS * 2, stream>>>(xbf, ckA, cbias, out);
}
